// MultiRelationalGATv2_35296041238628
// MI455X (gfx1250) — matrix operand order repaired and checked by reading, not compiled, not run on hardware
//
#include <hip/hip_runtime.h>
#include <cstddef>

// Problem constants (match reference)
#define NN   50000
#define INC  128
#define HH   8
#define CCH  16
#define EE   800000
#define ETOT (EE + NN)   // 850000 edges incl. self loops
#define RR   2
#define NT16 (NN / 16)   // 3125 M-tiles per GEMM

typedef __attribute__((ext_vector_type(16))) _Float16 v16h;
typedef __attribute__((ext_vector_type(8)))  _Float16 v8h;
typedef __attribute__((ext_vector_type(8)))  float    v8f;

// ---------------------------------------------------------------------------
// Convert x -> f16, and pre-swizzle the 4 weight matrices (Wl0,Wr0,Wl1,Wr1)
// into WMMA B-fragment layout:
//   w16[((g*4 + kt)*8 + nt)*512 + lane*16 + e] = W_g[k][col]
//   with n = lane&15, kh = lane>>4, k = kt*32 + (e<8 ? kh*8+e : 16+kh*8+e-8), col = nt*16 + n
// so the GEMM's per-lane B load is one contiguous 32-byte vector.
// ---------------------------------------------------------------------------
__global__ __launch_bounds__(256) void k_convert(
    const float* __restrict__ x, const float* __restrict__ Wl,
    const float* __restrict__ Wr, _Float16* __restrict__ x16,
    _Float16* __restrict__ w16)
{
  const int idx = blockIdx.x * 256 + (int)threadIdx.x;
  if (idx < NN * INC) x16[idx] = (_Float16)x[idx];
  if (idx < 4 * INC * INC) {
    const int e    = idx & 15;
    const int lane = (idx >> 4) & 31;
    const int nt   = (idx >> 9) & 7;
    const int kt   = (idx >> 12) & 3;
    const int g    = idx >> 14;
    const int n    = lane & 15;
    const int kh   = lane >> 4;
    const int k    = kt * 32 + ((e < 8) ? (kh * 8 + e) : (16 + kh * 8 + (e - 8)));
    const int col  = nt * 16 + n;
    const float* W = ((g & 1) ? Wr : Wl) + (g >> 1) * (INC * INC);
    w16[idx] = (_Float16)W[k * INC + col];
  }
}

// out = bias[0] + bias[1] (scatter target), den = 0. Runs every launch.
__global__ __launch_bounds__(256) void k_init(
    const float* __restrict__ bias, float* __restrict__ out,
    float* __restrict__ den)
{
  const int idx = blockIdx.x * 256 + (int)threadIdx.x;
  if (idx < NN * INC) out[idx] = bias[idx & 127] + bias[128 + (idx & 127)];
  if (idx < RR * NN * HH) den[idx] = 0.0f;
}

// ---------------------------------------------------------------------------
// 4 GEMMs [N x 128] @ [128 x 128] via v_wmma_f32_16x16x32_f16.
// One wave computes a 16(M) x 128(N) slab: 4 K-steps x 8 N-tiles = 32 WMMAs.
// xfeat layout: [g][node][128] with g = 2*r + (0=left/1=right transform).
// ---------------------------------------------------------------------------
__global__ __launch_bounds__(256) void k_gemm(
    const _Float16* __restrict__ x16, const _Float16* __restrict__ w16,
    const float* __restrict__ bl, const float* __restrict__ br,
    float* __restrict__ xfeat)
{
  const int wid  = (blockIdx.x * 256 + (int)threadIdx.x) >> 5;
  const int lane = (int)threadIdx.x & 31;
  const int g    = wid / NT16;
  const int mt   = wid - g * NT16;
  if (g >= 4) return;                 // uniform per wave; EXEC stays all-ones

  const int l15  = lane & 15;
  const int half = lane >> 4;
  const _Float16* xrow = x16 + (size_t)(mt * 16 + l15) * INC;
  const _Float16* wb   = w16 + (size_t)g * 16384 + (size_t)lane * 16;

  const v8f z = {0.f, 0.f, 0.f, 0.f, 0.f, 0.f, 0.f, 0.f};
  v8f acc[8];
#pragma unroll
  for (int n = 0; n < 8; ++n) acc[n] = z;

#pragma unroll
  for (int kt = 0; kt < 4; ++kt) {
    const int k0 = kt * 32;
    // ISA 16-bit A layout: lanes<16 take K {0..7,16..23}, lanes>=16 take
    // K {8..15,24..31} of this K-step -> two contiguous 16B chunks.
    v8h alo = *(const v8h*)(xrow + k0 + 8 * half);
    v8h ahi = *(const v8h*)(xrow + k0 + 16 + 8 * half);
    v16h a  = __builtin_shufflevector(alo, ahi,
                0, 1, 2, 3, 4, 5, 6, 7, 8, 9, 10, 11, 12, 13, 14, 15);
    const _Float16* wkt = wb + (size_t)kt * 4096;
#pragma unroll
    for (int nt = 0; nt < 8; ++nt) {
      v16h b = *(const v16h*)(wkt + (size_t)nt * 512);
      acc[nt] = __builtin_amdgcn_wmma_f32_16x16x32_f16(
          false, a, false, b, (short)0, acc[nt], false, false);
    }
  }

  const float* bv = ((g & 1) ? br : bl) + (g >> 1) * INC;
  float* og = xfeat + (size_t)g * NN * INC;
#pragma unroll
  for (int nt = 0; nt < 8; ++nt) {
    const int   col = nt * 16 + l15;
    const float bn  = bv[col];
#pragma unroll
    for (int r8 = 0; r8 < 8; ++r8) {
      const int row = mt * 16 + r8 + 8 * half;   // C/D: VGPR r -> M=r / r+8
      og[(size_t)row * INC + col] = acc[nt][r8] + bn;
    }
  }
}

// ---------------------------------------------------------------------------
// Pass A: per (relation, edge, head): score = att . lrelu(xl[src]+xr[dst]),
// ex = exp(score) (max-free softmax: scores ~N(0,1), exp is fp32-safe),
// store ex, atomic-accumulate denominator per (dst, head).
// ---------------------------------------------------------------------------
__global__ __launch_bounds__(256) void k_score(
    const int* __restrict__ ei0, const int* __restrict__ ei1,
    const float* __restrict__ xfeat, const float* __restrict__ att,
    float* __restrict__ exv, float* __restrict__ den)
{
  const int idx = blockIdx.x * 256 + (int)threadIdx.x;
  if (idx >= RR * ETOT * HH) return;
  const int h = idx & 7;
  const int t = idx >> 3;                  // r*ETOT + e
  const int r = (t >= ETOT) ? 1 : 0;
  const int e = t - r * ETOT;
  const int* ei = r ? ei1 : ei0;
  int src, dst;
  if (e < EE) { src = ei[e]; dst = ei[EE + e]; }
  else        { src = e - EE; dst = src; }

  const float4* pl = (const float4*)(xfeat + (size_t)(2 * r) * NN * INC +
                                     (size_t)src * INC + h * CCH);
  const float4* pr = (const float4*)(xfeat + (size_t)(2 * r + 1) * NN * INC +
                                     (size_t)dst * INC + h * CCH);
  const float4* pa = (const float4*)(att + (r * HH + h) * CCH);

  float s = 0.f;
#pragma unroll
  for (int q = 0; q < 4; ++q) {
    const float4 vl = pl[q], vr = pr[q], va = pa[q];
    float v0 = vl.x + vr.x; v0 = (v0 > 0.f) ? v0 : 0.2f * v0;
    float v1 = vl.y + vr.y; v1 = (v1 > 0.f) ? v1 : 0.2f * v1;
    float v2 = vl.z + vr.z; v2 = (v2 > 0.f) ? v2 : 0.2f * v2;
    float v3 = vl.w + vr.w; v3 = (v3 > 0.f) ? v3 : 0.2f * v3;
    s = fmaf(va.x, v0, s);
    s = fmaf(va.y, v1, s);
    s = fmaf(va.z, v2, s);
    s = fmaf(va.w, v3, s);
  }
  const float ex = __expf(s);
  exv[idx] = ex;
  atomicAdd(&den[(r * NN + dst) * HH + h], ex);
}

// ---------------------------------------------------------------------------
// Pass B: per (relation, edge, 4-channel group): alpha = ex/den[dst],
// out[dst, c] += alpha * xl[src, c] via fp32 atomics (L2-resident).
// ---------------------------------------------------------------------------
__global__ __launch_bounds__(256) void k_scatter(
    const int* __restrict__ ei0, const int* __restrict__ ei1,
    const float* __restrict__ xfeat, const float* __restrict__ exv,
    const float* __restrict__ den, float* __restrict__ out)
{
  const int idx = blockIdx.x * 256 + (int)threadIdx.x;
  if (idx >= RR * ETOT * 32) return;
  const int q = idx & 31;                  // channels q*4 .. q*4+3
  const int t = idx >> 5;                  // r*ETOT + e
  const int r = (t >= ETOT) ? 1 : 0;
  const int e = t - r * ETOT;
  const int* ei = r ? ei1 : ei0;
  int src, dst;
  if (e < EE) { src = ei[e]; dst = ei[EE + e]; }
  else        { src = e - EE; dst = src; }

  const int   h     = q >> 2;              // head = (q*4)/16
  const float alpha = exv[t * HH + h] / den[(r * NN + dst) * HH + h];
  const float4 xs = *(const float4*)(xfeat + (size_t)(2 * r) * NN * INC +
                                     (size_t)src * INC + q * 4);
  float* po = out + (size_t)dst * INC + q * 4;
  atomicAdd(po + 0, alpha * xs.x);
  atomicAdd(po + 1, alpha * xs.y);
  atomicAdd(po + 2, alpha * xs.z);
  atomicAdd(po + 3, alpha * xs.w);
}

// ---------------------------------------------------------------------------
extern "C" void kernel_launch(void* const* d_in, const int* in_sizes, int n_in,
                              void* d_out, int out_size, void* d_ws, size_t ws_size,
                              hipStream_t stream) {
  (void)in_sizes; (void)n_in; (void)out_size; (void)ws_size;

  const float* x    = (const float*)d_in[0];
  const int*   ei0  = (const int*)  d_in[1];
  const int*   ei1  = (const int*)  d_in[2];
  const float* Wl   = (const float*)d_in[3];
  const float* bl   = (const float*)d_in[4];
  const float* Wr   = (const float*)d_in[5];
  const float* br   = (const float*)d_in[6];
  const float* att  = (const float*)d_in[7];
  const float* bias = (const float*)d_in[8];
  float* out = (float*)d_out;

  // Workspace carve-up (~173 MB total, all 256B-aligned sizes)
  char* p = (char*)d_ws;
  _Float16* x16   = (_Float16*)p; p += (size_t)NN * INC * sizeof(_Float16);      // 12.8 MB
  _Float16* w16   = (_Float16*)p; p += (size_t)4 * INC * INC * sizeof(_Float16); // 128 KB
  float*    xfeat = (float*)p;    p += (size_t)4 * NN * INC * sizeof(float);     // 102.4 MB
  float*    exv   = (float*)p;    p += (size_t)RR * ETOT * HH * sizeof(float);   // 54.4 MB
  float*    den   = (float*)p;    p += (size_t)RR * NN * HH * sizeof(float);     // 3.2 MB

  const int tconv = NN * INC;                         // covers 4*128*128 too
  k_convert<<<(tconv + 255) / 256, 256, 0, stream>>>(x, Wl, Wr, x16, w16);
  k_init<<<(NN * INC + 255) / 256, 256, 0, stream>>>(bias, out, den);

  const int gemm_threads = 4 * NT16 * 32;             // one wave per (g, m-tile)
  k_gemm<<<(gemm_threads + 255) / 256, 256, 0, stream>>>(x16, w16, bl, br, xfeat);

  const int tA = RR * ETOT * HH;                      // 13.6M
  k_score<<<(tA + 255) / 256, 256, 0, stream>>>(ei0, ei1, xfeat, att, exv, den);

  const int tB = RR * ETOT * 32;                      // 54.4M
  k_scatter<<<(tB + 255) / 256, 256, 0, stream>>>(ei0, ei1, xfeat, exv, den, out);
}
